// SparseAttentionModule_43946105372911
// MI455X (gfx1250) — matrix operand order repaired and checked by reading, not compiled, not run on hardware
//
#include <hip/hip_runtime.h>
#include <hip/hip_bf16.h>

// ---------------------------------------------------------------------------
// Problem constants (match reference)
// ---------------------------------------------------------------------------
constexpr int S_  = 1024;   // seq len
constexpr int D_  = 512;    // d_model
constexpr int H_  = 8;      // heads
constexpr int DK_ = 64;     // head dim
constexpr int B_  = 4;      // batch
constexpr int PD_ = 64;     // pattern dim
constexpr int NP_ = 8;      // num patterns
constexpr int KTOP_ = 204;  // int(1024*(1-0.8))
constexpr float NEGV = -1e9f;

typedef _Float16 f16t;
typedef __attribute__((ext_vector_type(16))) _Float16 v16h;
typedef __attribute__((ext_vector_type(8)))  float    v8f;

// ---------------------------------------------------------------------------
// wave32 helpers
// ---------------------------------------------------------------------------
__device__ inline int wsum_i(int v) {
#pragma unroll
  for (int o = 16; o > 0; o >>= 1) v += __shfl_xor(v, o, 32);
  return v;
}

__device__ inline unsigned fkey(float f) {
  unsigned u = __float_as_uint(f);
  return (u & 0x80000000u) ? ~u : (u | 0x80000000u);  // monotonic float->uint
}

// ---------------------------------------------------------------------------
// Histogram radix select (k-th largest of 1024 keys, 4 byte passes).
// One wave per block; lane l owns keys of elements [l*32, l*32+32).
// LDS histogram + ds_add atomics; digit located via wave suffix-scan.
// ---------------------------------------------------------------------------
__device__ unsigned radix_kth_hist(const unsigned keys[32], int lane, int k,
                                   int* gt_out)
{
  __shared__ unsigned hist[258];   // 256 bins + {digit, kk'} broadcast slots
  unsigned prefix = 0u;
  int kk = k;
  for (int pass = 3; pass >= 0; --pass) {
    const int sh = pass * 8;
    const unsigned pmask = (pass == 3) ? 0u : (0xFFFFFFFFu << (sh + 8));
#pragma unroll
    for (int t = 0; t < 8; ++t) hist[lane * 8 + t] = 0u;
    __syncthreads();
#pragma unroll
    for (int t = 0; t < 32; ++t) {
      unsigned key = keys[t];
      if ((key & pmask) == prefix)
        atomicAdd(&hist[(key >> sh) & 255u], 1u);
    }
    __syncthreads();
    unsigned bins[8];
    int lsum = 0;
#pragma unroll
    for (int t = 0; t < 8; ++t) { bins[t] = hist[lane * 8 + t]; lsum += (int)bins[t]; }
    // inclusive suffix-sum over lanes (lane owns digits lane*8..lane*8+7)
    int suf = lsum;
#pragma unroll
    for (int o = 1; o < 32; o <<= 1) {
      int u = __shfl_down(suf, o, 32);
      if (lane + o < 32) suf += u;
    }
    int run = suf - lsum;            // count of candidate keys with digit in higher lanes
#pragma unroll
    for (int t = 7; t >= 0; --t) {   // descend digits within this lane
      int nrun = run + (int)bins[t];
      if (run < kk && nrun >= kk) {  // exactly one (lane,t) crosses
        hist[256] = (unsigned)(lane * 8 + t);
        hist[257] = (unsigned)(kk - run);
      }
      run = nrun;
    }
    __syncthreads();
    prefix |= hist[256] << sh;
    kk = (int)hist[257];
    __syncthreads();
  }
  int g = 0;
#pragma unroll
  for (int t = 0; t < 32; ++t) g += (keys[t] > prefix) ? 1 : 0;
  *gt_out = wsum_i(g);
  return prefix;
}

// index-ordered tie marking helper: returns per-element take decisions as bits
__device__ unsigned mark_topk(const unsigned keys[32], int lane, unsigned thr, int rem)
{
  int eq = 0;
#pragma unroll
  for (int t = 0; t < 32; ++t) eq += (keys[t] == thr) ? 1 : 0;
  int pre = eq;
#pragma unroll
  for (int o = 1; o < 32; o <<= 1) {
    int u = __shfl_up(pre, o, 32);
    if (lane >= o) pre += u;
  }
  int excl = pre - eq, seen = 0;
  unsigned word = 0u;
#pragma unroll
  for (int t = 0; t < 32; ++t) {
    unsigned key = keys[t];
    bool take = false;
    if (key > thr) take = true;
    else if (key == thr) { if (excl + seen < rem) take = true; ++seen; }
    if (take) word |= 1u << t;
  }
  return word;
}

// threefry2x32-style deterministic PRNG (key = 42); returns 64 random bits
__device__ inline unsigned rotl32(unsigned x, int r) { return (x << r) | (x >> (32 - r)); }
__device__ inline void threefry_pair(unsigned c0, unsigned c1, unsigned& o0, unsigned& o1) {
  const unsigned ks0 = 0u, ks1 = 42u;
  const unsigned ks2 = 0x1BD11BDAu ^ ks0 ^ ks1;
  unsigned x0 = c0 + ks0;
  unsigned x1 = c1 + ks1;
  const int ra[4] = {13, 15, 26, 6};
  const int rb[4] = {17, 29, 16, 24};
#pragma unroll
  for (int i = 0; i < 4; ++i) { x0 += x1; x1 = rotl32(x1, ra[i]); x1 ^= x0; }
  x0 += ks1; x1 += ks2 + 1u;
#pragma unroll
  for (int i = 0; i < 4; ++i) { x0 += x1; x1 = rotl32(x1, rb[i]); x1 ^= x0; }
  x0 += ks2; x1 += ks0 + 2u;
#pragma unroll
  for (int i = 0; i < 4; ++i) { x0 += x1; x1 = rotl32(x1, ra[i]); x1 ^= x0; }
  x0 += ks0; x1 += ks1 + 3u;
#pragma unroll
  for (int i = 0; i < 4; ++i) { x0 += x1; x1 = rotl32(x1, rb[i]); x1 ^= x0; }
  x0 += ks1; x1 += ks2 + 4u;
#pragma unroll
  for (int i = 0; i < 4; ++i) { x0 += x1; x1 = rotl32(x1, ra[i]); x1 ^= x0; }
  x0 += ks2; x1 += ks0 + 5u;
  o0 = x0; o1 = x1;
}

// ---------------------------------------------------------------------------
// Generic tiled WMMA GEMM:  C = scale * (A @ op(B)) + bias
//   A: [M,K] f32 row-major.  TRANSB=0: B [K,N] row-major.  TRANSB=1: B [N,K].
//   Block tile 32x64, 8 waves, each wave one 16x16 WMMA subtile (2x4 grid).
//   All staging loads are contiguous float4 (global_load_b128).
//   grid = (N/64, M/32, batch)
// ---------------------------------------------------------------------------
template <int TRANSB>
__global__ __launch_bounds__(256) void wmma_gemm_f32(
    const float* __restrict__ A, int lda, long long sA,
    const float* __restrict__ Bm, int ldb, long long sB,
    const float* __restrict__ bias,
    float* __restrict__ C, int ldc, long long sC,
    int K, float scale)
{
  __shared__ f16t As[32 * 34];   // 32x32 A tile (+pad)
  __shared__ f16t Bs[32 * 66];   // 32x64 B tile [k][n] (+pad)

  const int tid  = threadIdx.x;
  const int lane = tid & 31;
  const int wid  = tid >> 5;
  const int wr   = wid >> 2;     // 0..1  subtile row
  const int wc   = wid & 3;      // 0..3  subtile col
  const int Nb = blockIdx.x * 64;
  const int Mb = blockIdx.y * 32;
  const float* Ab = A  + (long long)blockIdx.z * sA;
  const float* Bb = Bm + (long long)blockIdx.z * sB;
  float*       Cb = C  + (long long)blockIdx.z * sC;

  const int ar = tid >> 3;           // A tile row 0..31
  const int ak = (tid & 7) * 4;      // A tile col chunk (float4)
  // TRANSB=0: thread owns (k=tid>>3, 8 consecutive n)
  const int bk0 = tid >> 3;
  const int bn0 = (tid & 7) * 8;
  // TRANSB=1: thread owns (n=tid>>2, 8 consecutive k)  -> contiguous loads
  const int bn1 = tid >> 2;
  const int bk1 = (tid & 3) * 8;

  const int m     = lane & 15;
  const int khalf = lane >> 4;

  v8f acc = {};

  for (int kk = 0; kk < K; kk += 32) {
    {
      float4 x = *(const float4*)&Ab[(size_t)(Mb + ar) * lda + kk + ak];
      As[ar * 34 + ak + 0] = (f16t)x.x;
      As[ar * 34 + ak + 1] = (f16t)x.y;
      As[ar * 34 + ak + 2] = (f16t)x.z;
      As[ar * 34 + ak + 3] = (f16t)x.w;
    }
    if (TRANSB) {
      const float* bp = &Bb[(size_t)(Nb + bn1) * ldb + kk + bk1];
      float4 x0 = *(const float4*)bp;
      float4 x1 = *(const float4*)(bp + 4);
      Bs[(bk1 + 0) * 66 + bn1] = (f16t)x0.x;
      Bs[(bk1 + 1) * 66 + bn1] = (f16t)x0.y;
      Bs[(bk1 + 2) * 66 + bn1] = (f16t)x0.z;
      Bs[(bk1 + 3) * 66 + bn1] = (f16t)x0.w;
      Bs[(bk1 + 4) * 66 + bn1] = (f16t)x1.x;
      Bs[(bk1 + 5) * 66 + bn1] = (f16t)x1.y;
      Bs[(bk1 + 6) * 66 + bn1] = (f16t)x1.z;
      Bs[(bk1 + 7) * 66 + bn1] = (f16t)x1.w;
    } else {
      const float* bp = &Bb[(size_t)(kk + bk0) * ldb + Nb + bn0];
      float4 x0 = *(const float4*)bp;
      float4 x1 = *(const float4*)(bp + 4);
      Bs[bk0 * 66 + bn0 + 0] = (f16t)x0.x;
      Bs[bk0 * 66 + bn0 + 1] = (f16t)x0.y;
      Bs[bk0 * 66 + bn0 + 2] = (f16t)x0.z;
      Bs[bk0 * 66 + bn0 + 3] = (f16t)x0.w;
      Bs[bk0 * 66 + bn0 + 4] = (f16t)x1.x;
      Bs[bk0 * 66 + bn0 + 5] = (f16t)x1.y;
      Bs[bk0 * 66 + bn0 + 6] = (f16t)x1.z;
      Bs[bk0 * 66 + bn0 + 7] = (f16t)x1.w;
    }
    if (kk + 32 < K) {  // gfx1250: global_prefetch_b8 of the next A tile
      __builtin_prefetch(&Ab[(size_t)(Mb + ar) * lda + kk + 32 + ak], 0, 1);
    }
    __syncthreads();

    // 16-bit A / B fragments, contiguous k = 16*khalf + s (wave32)
    v16h af, bf;
#pragma unroll
    for (int v = 0; v < 8; ++v) {
      int ka = khalf * 16 + 2 * v;
      af[2 * v]     = As[(wr * 16 + m) * 34 + ka];
      af[2 * v + 1] = As[(wr * 16 + m) * 34 + ka + 1];
      int kb = khalf * 16 + 2 * v;
      bf[2 * v]     = Bs[kb * 66 + wc * 16 + m];
      bf[2 * v + 1] = Bs[(kb + 1) * 66 + wc * 16 + m];
    }
    acc = __builtin_amdgcn_wmma_f32_16x16x32_f16(false, af, false, bf,
                                                 (short)0, acc, false, false);
    __syncthreads();
  }

  const int rbase = khalf * 8;       // C/D layout: lanes16-31 hold M+8
#pragma unroll
  for (int v = 0; v < 8; ++v) {
    int row = Mb + wr * 16 + rbase + v;
    int col = Nb + wc * 16 + m;
    float r = acc[v] * scale;
    if (bias) r += bias[col];
    Cb[(size_t)row * ldc + col] = r;
  }
}

// ---------------------------------------------------------------------------
// Pattern learner: enc = tanh(relu(q@pw1+pb1)@pw2+pb2); softmax(enc@bankT/8)*gate
// ---------------------------------------------------------------------------
__global__ __launch_bounds__(64) void pattern_kernel(
    const float* __restrict__ query,
    const float* __restrict__ pw1, const float* __restrict__ pb1,
    const float* __restrict__ pw2, const float* __restrict__ pb2,
    const float* __restrict__ bank, const float* __restrict__ gate,
    float* __restrict__ outp)
{
  __shared__ float q[512];
  __shared__ float h1[64];
  __shared__ float h2[64];
  __shared__ float ps[8];
  const int t = threadIdx.x;
  const int row = blockIdx.x;
  const float* qr = query + (size_t)row * D_;
  for (int e = 0; e < 8; ++e) q[t * 8 + e] = qr[t * 8 + e];
  __syncthreads();
  float a1 = pb1[t];
  for (int k = 0; k < 512; ++k) a1 += q[k] * pw1[k * PD_ + t];
  h1[t] = fmaxf(a1, 0.f);
  __syncthreads();
  float a2 = pb2[t];
  for (int k = 0; k < 64; ++k) a2 += h1[k] * pw2[k * PD_ + t];
  h2[t] = tanhf(a2);
  __syncthreads();
  if (t < 8) {
    float s = 0.f;
    for (int k = 0; k < 64; ++k) s += h2[k] * bank[t * PD_ + k];
    ps[t] = s * 0.125f;
  }
  __syncthreads();
  if (t == 0) {
    float mx = ps[0];
    for (int p = 1; p < 8; ++p) mx = fmaxf(mx, ps[p]);
    float e[8], sum = 0.f;
    for (int p = 0; p < 8; ++p) { e[p] = __expf(ps[p] - mx); sum += e[p]; }
    float inv = 1.f / sum;
    for (int p = 0; p < 8; ++p) outp[(size_t)row * NP_ + p] = e[p] * inv * gate[p];
  }
}

// ---------------------------------------------------------------------------
// Mask stage 1: per (b,i) row of scores3, top-204 -> bitmask words. 1 wave/row.
// ---------------------------------------------------------------------------
__global__ __launch_bounds__(32) void mask_g3_kernel(
    const float* __restrict__ s3, unsigned* __restrict__ g3w)
{
  const int lane = threadIdx.x;
  const int row  = blockIdx.x;                 // b*S + i
  const float* src = s3 + (size_t)row * S_;
  unsigned keys[32];
#pragma unroll
  for (int t = 0; t < 32; ++t) keys[t] = fkey(src[lane * 32 + t]);
  int gt;
  unsigned thr  = radix_kth_hist(keys, lane, KTOP_, &gt);
  unsigned word = mark_topk(keys, lane, thr, KTOP_ - gt);
  g3w[(size_t)row * 32 + lane] = word;
}

// ---------------------------------------------------------------------------
// Mask stage 2: combined = local | g3 | rnd; masked = s3*combined (excluded->0),
// top-204 again, write 0/1 float rows. 1 wave per (b,h,i).
// ---------------------------------------------------------------------------
__global__ __launch_bounds__(32) void mask_final_kernel(
    const float* __restrict__ s3, const unsigned* __restrict__ g3w,
    float* __restrict__ outmask)
{
  const int lane = threadIdx.x;
  const int bid  = blockIdx.x;         // (b*H + h)*S + i
  const int i    = bid & (S_ - 1);
  const int bh   = bid >> 10;
  const int b    = bh >> 3;
  const float* src = s3 + ((size_t)(b * S_) + i) * S_;
  unsigned gw = g3w[((size_t)(b * S_) + i) * 32 + lane];

  unsigned keys[32];
#pragma unroll
  for (int q = 0; q < 4; ++q) {
    unsigned r0, r1;
    threefry_pair(((unsigned)bid << 7) | (unsigned)(lane * 4 + q), 0u, r0, r1);
#pragma unroll
    for (int tt = 0; tt < 8; ++tt) {
      int t = q * 8 + tt;
      int j = lane * 32 + t;
      unsigned byt = ((tt < 4) ? (r0 >> (8 * tt)) : (r1 >> (8 * (tt - 4)))) & 255u;
      bool rn  = byt < 51u;                   // p = 51/256 ~= 0.2
      bool loc = (j >= i - 32) && (j < i + 32);
      bool g3  = (gw >> t) & 1u;
      float v  = src[j];
      keys[t]  = fkey((loc || g3 || rn) ? v : 0.f);
    }
  }
  int gt;
  unsigned thr  = radix_kth_hist(keys, lane, KTOP_, &gt);
  unsigned word = mark_topk(keys, lane, thr, KTOP_ - gt);
  float* orow = outmask + (size_t)bid * S_;
#pragma unroll
  for (int t = 0; t < 32; ++t)
    orow[lane * 32 + t] = ((word >> t) & 1u) ? 1.f : 0.f;
}

// ---------------------------------------------------------------------------
// Per-head masked attention. One wave owns 16 query rows of one (b,h):
//  WMMA Q@K^T (2 k-steps over DK=64) -> mask -> LDS softmax over 1024 ->
//  WMMA P@V (32 k-steps x 4 N-subtiles, V tile staged in LDS).
// Dynamic LDS: 64KB score strip + 4.25KB V tile.
// ---------------------------------------------------------------------------
__global__ __launch_bounds__(32) void attn_head_kernel(
    const float* __restrict__ Qf, const float* __restrict__ Kf,
    const float* __restrict__ Vf, const float* __restrict__ mask,
    float* __restrict__ outws)
{
  extern __shared__ unsigned char smem_raw[];
  float* sc = (float*)smem_raw;                              // 16 x 1024 f32
  f16t*  Vs = (f16t*)(smem_raw + 16 * 1024 * sizeof(float)); // 32 x 68 f16

  const int lane = threadIdx.x;
  const int bid  = blockIdx.x;            // (b*H + h)*64 + it
  const int it   = bid & 63;
  const int bh   = bid >> 6;
  const int h    = bh & 7;
  const int b    = bh >> 3;
  const int i0   = it * 16;

  const int m     = lane & 15;
  const int khalf = lane >> 4;

  // Q fragments for this 16-row strip (k 0..31 and 32..63 of the head slice)
  v16h aq0, aq1;
  {
    const float* qr = Qf + ((size_t)(b * S_) + i0 + m) * D_ + h * DK_;
#pragma unroll
    for (int v = 0; v < 8; ++v) {
      int ka = khalf * 16 + 2 * v;
      float2 x0 = *(const float2*)&qr[ka];
      float2 x1 = *(const float2*)&qr[ka + 32];
      aq0[2 * v]     = (f16t)x0.x;
      aq0[2 * v + 1] = (f16t)x0.y;
      aq1[2 * v]     = (f16t)x1.x;
      aq1[2 * v + 1] = (f16t)x1.y;
    }
  }

  const float* mrow = mask + (((size_t)(b * H_ + h) * S_) + i0) * S_;

  for (int jt = 0; jt < 64; ++jt) {
    v16h bk0, bk1;  // B fragment: lane = n (key index jt*16+n), slots pack k
    const float* kr = Kf + ((size_t)(b * S_) + jt * 16 + m) * D_ + h * DK_;
#pragma unroll
    for (int v = 0; v < 8; ++v) {
      int kb = khalf * 16 + 2 * v;
      float2 x0 = *(const float2*)&kr[kb];
      float2 x1 = *(const float2*)&kr[kb + 32];
      bk0[2 * v]     = (f16t)x0.x;
      bk0[2 * v + 1] = (f16t)x0.y;
      bk1[2 * v]     = (f16t)x1.x;
      bk1[2 * v + 1] = (f16t)x1.y;
    }
    v8f acc = {};
    acc = __builtin_amdgcn_wmma_f32_16x16x32_f16(false, aq0, false, bk0, (short)0, acc, false, false);
    acc = __builtin_amdgcn_wmma_f32_16x16x32_f16(false, aq1, false, bk1, (short)0, acc, false, false);
#pragma unroll
    for (int v = 0; v < 8; ++v) {
      int ml = khalf * 8 + v;
      int j  = jt * 16 + m;
      float mv = mrow[(size_t)ml * S_ + j];
      sc[ml * 1024 + j] = (mv > 0.f) ? acc[v] * 0.125f : NEGV;
    }
  }
  __syncthreads();

  // softmax over each of the 16 rows (wave-cooperative)
  for (int r = 0; r < 16; ++r) {
    float mx = -3.0e38f;
    for (int t = 0; t < 32; ++t) mx = fmaxf(mx, sc[r * 1024 + lane + t * 32]);
#pragma unroll
    for (int o = 16; o > 0; o >>= 1) mx = fmaxf(mx, __shfl_xor(mx, o, 32));
    float sum = 0.f;
    for (int t = 0; t < 32; ++t) {
      int idx = r * 1024 + lane + t * 32;
      float e = __expf(sc[idx] - mx);
      sc[idx] = e;
      sum += e;
    }
#pragma unroll
    for (int o = 16; o > 0; o >>= 1) sum += __shfl_xor(sum, o, 32);
    float inv = 1.0f / sum;
    for (int t = 0; t < 32; ++t) sc[r * 1024 + lane + t * 32] *= inv;
  }

  // P @ V : 16x1024 @ 1024x64 -> 16x64, V tile staged in LDS per k-step
  v8f ao[4] = {v8f{}, v8f{}, v8f{}, v8f{}};
  for (int kt = 0; kt < 32; ++kt) {
    __syncthreads();
    {
      const float* vrow = Vf + ((size_t)(b * S_) + kt * 32 + lane) * D_ + h * DK_;
#pragma unroll
      for (int q = 0; q < 16; ++q) {
        float4 x = *(const float4*)&vrow[q * 4];
        Vs[lane * 68 + q * 4 + 0] = (f16t)x.x;
        Vs[lane * 68 + q * 4 + 1] = (f16t)x.y;
        Vs[lane * 68 + q * 4 + 2] = (f16t)x.z;
        Vs[lane * 68 + q * 4 + 3] = (f16t)x.w;
      }
    }
    __syncthreads();
    v16h ap;
#pragma unroll
    for (int v = 0; v < 8; ++v) {
      int ka = kt * 32 + khalf * 16 + 2 * v;
      float2 x = *(const float2*)&sc[m * 1024 + ka];
      ap[2 * v]     = (f16t)x.x;
      ap[2 * v + 1] = (f16t)x.y;
    }
#pragma unroll
    for (int nt = 0; nt < 4; ++nt) {
      v16h bv;
#pragma unroll
      for (int v = 0; v < 8; ++v) {
        int kb = khalf * 16 + 2 * v;
        bv[2 * v]     = Vs[kb * 68 + nt * 16 + m];
        bv[2 * v + 1] = Vs[(kb + 1) * 68 + nt * 16 + m];
      }
      ao[nt] = __builtin_amdgcn_wmma_f32_16x16x32_f16(false, ap, false, bv,
                                                      (short)0, ao[nt], false, false);
    }
  }

#pragma unroll
  for (int nt = 0; nt < 4; ++nt) {
#pragma unroll
    for (int v = 0; v < 8; ++v) {
      int row = i0 + khalf * 8 + v;
      int col = h * DK_ + nt * 16 + m;
      outws[((size_t)(b * S_) + row) * D_ + col] = ao[nt][v];
    }
  }
}

// ---------------------------------------------------------------------------
// Launch
// ---------------------------------------------------------------------------
extern "C" void kernel_launch(void* const* d_in, const int* in_sizes, int n_in,
                              void* d_out, int out_size, void* d_ws, size_t ws_size,
                              hipStream_t stream)
{
  (void)in_sizes; (void)n_in; (void)out_size; (void)ws_size;

  const float* query = (const float*)d_in[0];
  const float* key   = (const float*)d_in[1];
  const float* value = (const float*)d_in[2];
  const float* wq = (const float*)d_in[3];   const float* bq = (const float*)d_in[4];
  const float* wk = (const float*)d_in[5];   const float* bk = (const float*)d_in[6];
  const float* wv = (const float*)d_in[7];   const float* bv = (const float*)d_in[8];
  const float* wo = (const float*)d_in[9];   const float* bo = (const float*)d_in[10];
  const float* pw1 = (const float*)d_in[11]; const float* pb1 = (const float*)d_in[12];
  const float* pw2 = (const float*)d_in[13]; const float* pb2 = (const float*)d_in[14];
  const float* bank = (const float*)d_in[15];
  const float* gate = (const float*)d_in[16];

  // workspace layout (floats): Qf | Kf | Vf | scores3 | attn_out | g3 bitmask
  float* ws = (float*)d_ws;
  float*    Qf  = ws;                         // 4096*512
  float*    Kf  = ws + 2097152;
  float*    Vf  = ws + 4194304;
  float*    s3  = ws + 6291456;               // 4*1024*1024
  float*    aow = ws + 10485760;              // 4096*512
  unsigned* g3w = (unsigned*)(ws + 12582912); // 4*1024*32 words

  float* out_att  = (float*)d_out;            // (B,S,D)
  float* out_mask = out_att + 2097152;        // (B,H,S,S)
  float* out_pat  = out_att + 35651584;       // (B,S,NP)

  dim3 blk(256);
  dim3 gproj(512 / 64, 4096 / 32, 1);

  // Q/K/V projections: [4096,512] @ [512,512] + bias
  wmma_gemm_f32<0><<<gproj, blk, 0, stream>>>(query, 512, 0, wq, 512, 0, bq, Qf, 512, 0, 512, 1.0f);
  wmma_gemm_f32<0><<<gproj, blk, 0, stream>>>(key,   512, 0, wk, 512, 0, bk, Kf, 512, 0, 512, 1.0f);
  wmma_gemm_f32<0><<<gproj, blk, 0, stream>>>(value, 512, 0, wv, 512, 0, bv, Vf, 512, 0, 512, 1.0f);

  // pattern learner (independent of above)
  pattern_kernel<<<dim3(B_ * S_), dim3(64), 0, stream>>>(query, pw1, pb1, pw2, pb2,
                                                         bank, gate, out_pat);

  // scores3 = Qf @ Kf^T / 8, batched over B
  dim3 gs3(1024 / 64, 1024 / 32, B_);
  wmma_gemm_f32<1><<<gs3, blk, 0, stream>>>(Qf, 512, 524288LL, Kf, 512, 524288LL,
                                            nullptr, s3, 1024, 1048576LL, 512, 0.125f);

  // mask generation
  mask_g3_kernel<<<dim3(B_ * S_), dim3(32), 0, stream>>>(s3, g3w);
  mask_final_kernel<<<dim3(B_ * H_ * S_), dim3(32), 0, stream>>>(s3, g3w, out_mask);

  // masked multi-head attention
  size_t attn_lds = 16 * 1024 * sizeof(float) + 32 * 68 * sizeof(f16t);
  attn_head_kernel<<<dim3(B_ * H_ * (S_ / 16)), dim3(32), attn_lds, stream>>>(
      Qf, Kf, Vf, out_mask, aow);

  // output projection
  wmma_gemm_f32<0><<<gproj, blk, 0, stream>>>(aow, 512, 0, wo, 512, 0, bo,
                                              out_att, 512, 0, 512, 1.0f);
}
